// AttentionBlock_80315888435976
// MI455X (gfx1250) — matrix operand order repaired and checked by reading, not compiled, not run on hardware
//
#include <hip/hip_runtime.h>
#include <hip/hip_bf16.h>

// ---------------------------------------------------------------------------
// AttentionBlock on MI455X (gfx1250): f16 WMMA pipeline, f32 accumulate.
//   B=32, C=512, H=W=32 -> N=1024, GROUPS=16
// GEMMs: block tile 128x64 (8 waves, wave32), wave tile 32x32 ->
// 4x v_wmma_f32_16x16x32_f16 per K-chunk, double-buffered LDS pipeline.
// LDS tiles are fragment-major (ldsA[m][k], ldsB[n][k]) so every WMMA
// fragment is two contiguous ds_load_b128 per lane.
// ---------------------------------------------------------------------------

typedef __attribute__((ext_vector_type(16))) _Float16 v16h;
typedef __attribute__((ext_vector_type(8)))  float    v8f;

union Frag16 { v16h v; _Float16 h[16]; unsigned u[8]; uint4 q[2]; };
union H8     { uint4 q; _Float16 h[8]; };

// ---------------------------------------------------------------------------
__global__ __launch_bounds__(256)
void cvt_f16_k(const float* __restrict__ src, _Float16* __restrict__ dst, int n)
{
    int i = blockIdx.x * 256 + threadIdx.x;
    if (i < n) dst[i] = (_Float16)src[i];
}

// ---------------------------------------------------------------------------
// GroupNorm: one block per (batch, group); contiguous 32ch x 1024px slab.
// ---------------------------------------------------------------------------
__global__ __launch_bounds__(256)
void groupnorm_k(const float* __restrict__ x, const float* __restrict__ gamma,
                 const float* __restrict__ beta, _Float16* __restrict__ hn)
{
    __shared__ float red[256];
    __shared__ float red2[256];
    const int    blk  = blockIdx.x;          // b*16 + grp
    const int    grp  = blk & 15;
    const size_t base = (size_t)blk * (32 * 1024);

    float s = 0.f, s2 = 0.f;
    for (int i = threadIdx.x; i < 32 * 1024; i += 256) {
        float v = x[base + i];
        s += v; s2 += v * v;
    }
    red[threadIdx.x] = s; red2[threadIdx.x] = s2;
    __syncthreads();
    for (int o = 128; o > 0; o >>= 1) {
        if (threadIdx.x < o) {
            red[threadIdx.x]  += red[threadIdx.x + o];
            red2[threadIdx.x] += red2[threadIdx.x + o];
        }
        __syncthreads();
    }
    const float inv  = 1.0f / 32768.0f;
    const float mu   = red[0] * inv;
    const float var  = red2[0] * inv - mu * mu;
    const float rstd = rsqrtf(var + 1e-6f);

    for (int i = threadIdx.x; i < 32 * 1024; i += 256) {
        int   c = grp * 32 + (i >> 10);
        float v = (x[base + i] - mu) * rstd * gamma[c] + beta[c];
        hn[base + i] = (_Float16)v;
    }
}

// ---------------------------------------------------------------------------
// Tiled WMMA GEMM:  D[b] = A[b] (MxK) * B[b] (KxN)   (f16 in, f32 accum)
//   TA: A stored [K][M]    TB: B stored [N][K]
//   OMODE 0: f16 out   1: f32 out * scale   2: f32 out + residual
//   BIASM: add bias[m] before store
// ---------------------------------------------------------------------------
template<bool TA, bool TB, int OMODE, bool BIASM>
__global__ __launch_bounds__(256)
void gemm_wmma(const _Float16* __restrict__ A,  long long sAb,
               const _Float16* __restrict__ Bm, long long sBb,
               void* __restrict__ Out,          long long sOb,
               const float* __restrict__ bias,
               const float* __restrict__ res,
               int M, int N, int K, float scale)
{
    constexpr int LDA = 40;   // halves; 80B rows -> 16B aligned
    constexpr int LDB = 40;
    __shared__ _Float16 ldsA[2][128 * LDA];   // [m][k]  m:0..127 k:0..31
    __shared__ _Float16 ldsB[2][64 * LDB];    // [n][k]  n:0..63  k:0..31

    const int bz = blockIdx.z;
    const int M0 = blockIdx.y * 128;
    const int N0 = blockIdx.x * 64;
    const _Float16* Ab = A  + (size_t)bz * sAb;
    const _Float16* Bb = Bm + (size_t)bz * sBb;

    const int tid  = threadIdx.x;
    const int lane = tid & 31;
    const int wid  = tid >> 5;
    const int wm   = (wid & 3) * 32;   // wave M offset in tile
    const int wn   = (wid >> 2) * 32;  // wave N offset in tile
    const int l16  = lane & 15;
    const int lh   = lane >> 4;

    // per-thread staging coordinates
    const int rA  = tid >> 2, cA  = (tid & 3) * 8;   // !TA: 2 rows (rA, rA+64)
    const int rAt = tid >> 4, cAt = (tid & 15) * 8;  //  TA: k rows rAt, rAt+16
    const int rB  = tid >> 3, cB  = (tid & 7) * 8;   // !TB: k row,  n col
    const int rBt = tid >> 2, cBt = (tid & 3) * 8;   //  TB: n row,  k col

    // global -> registers (issued early; latency overlaps WMMA of prev chunk)
    auto gload = [&](int kb, H8& t0, H8& t1, H8& t2) {
        if constexpr (!TA) {
            t0.q = *(const uint4*)&Ab[(size_t)(M0 + rA) * K + kb + cA];
            t1.q = *(const uint4*)&Ab[(size_t)(M0 + rA + 64) * K + kb + cA];
        } else {
            t0.q = *(const uint4*)&Ab[(size_t)(kb + rAt) * M + M0 + cAt];
            t1.q = *(const uint4*)&Ab[(size_t)(kb + rAt + 16) * M + M0 + cAt];
        }
        if constexpr (!TB)
            t2.q = *(const uint4*)&Bb[(size_t)(kb + rB) * N + N0 + cB];
        else
            t2.q = *(const uint4*)&Bb[(size_t)(N0 + rBt) * K + kb + cBt];
    };
    // registers -> LDS (fragment-major layout)
    auto sstore = [&](int buf, const H8& t0, const H8& t1, const H8& t2) {
        _Float16* a = ldsA[buf];
        _Float16* b = ldsB[buf];
        if constexpr (!TA) {
            *(uint4*)&a[rA * LDA + cA]        = t0.q;
            *(uint4*)&a[(rA + 64) * LDA + cA] = t1.q;
        } else {   // transpose during LDS write
            #pragma unroll
            for (int u = 0; u < 8; ++u) a[(cAt + u) * LDA + rAt]      = t0.h[u];
            #pragma unroll
            for (int u = 0; u < 8; ++u) a[(cAt + u) * LDA + rAt + 16] = t1.h[u];
        }
        if constexpr (!TB) {
            #pragma unroll
            for (int u = 0; u < 8; ++u) b[(cB + u) * LDB + rB] = t2.h[u];
        } else {
            *(uint4*)&b[rBt * LDB + cBt] = t2.q;
        }
    };

    v8f acc00 = {}, acc01 = {}, acc10 = {}, acc11 = {};

    const int steps = K >> 5;
    H8 ta0, ta1, tb2;
    gload(0, ta0, ta1, tb2);
    sstore(0, ta0, ta1, tb2);
    __syncthreads();

    for (int i = 0; i < steps; ++i) {
        const int  buf  = i & 1;
        const bool more = (i + 1) < steps;
        if (more) gload((i + 1) << 5, ta0, ta1, tb2);

        // ---- fragments: 2x b128 per fragment per lane ----
        Frag16 a0, a1, b0, b1;
        {
            const _Float16* p0 = &ldsA[buf][(wm + l16) * LDA];
            a0.q[0] = *(const uint4*)(p0 + lh * 8);
            a0.q[1] = *(const uint4*)(p0 + 16 + lh * 8);
            const _Float16* p1 = &ldsA[buf][(wm + 16 + l16) * LDA];
            a1.q[0] = *(const uint4*)(p1 + lh * 8);
            a1.q[1] = *(const uint4*)(p1 + 16 + lh * 8);
            const _Float16* p2 = &ldsB[buf][(wn + l16) * LDB + lh * 8];
            b0.q[0] = *(const uint4*)(p2);
            b0.q[1] = *(const uint4*)(p2 + 16);
            const _Float16* p3 = &ldsB[buf][(wn + 16 + l16) * LDB + lh * 8];
            b1.q[0] = *(const uint4*)(p3);
            b1.q[1] = *(const uint4*)(p3 + 16);
        }

        if (more) sstore(buf ^ 1, ta0, ta1, tb2);   // fill other buffer

        acc00 = __builtin_amdgcn_wmma_f32_16x16x32_f16(false, a0.v, false, b0.v,
                                                       (short)0, acc00, false, false);
        acc01 = __builtin_amdgcn_wmma_f32_16x16x32_f16(false, a0.v, false, b1.v,
                                                       (short)0, acc01, false, false);
        acc10 = __builtin_amdgcn_wmma_f32_16x16x32_f16(false, a1.v, false, b0.v,
                                                       (short)0, acc10, false, false);
        acc11 = __builtin_amdgcn_wmma_f32_16x16x32_f16(false, a1.v, false, b1.v,
                                                       (short)0, acc11, false, false);
        __syncthreads();   // single barrier per chunk (double buffer)
    }

    // ---- store: D layout lane l -> col N=l%16, VGPR r -> row M=r+8*(l/16)
    const int n0 = N0 + wn + l16;
    #pragma unroll
    for (int r = 0; r < 8; ++r) {
        int m0 = M0 + wm + 8 * lh + r;
        int m1 = m0 + 16;
        float v00 = acc00[r], v01 = acc01[r], v10 = acc10[r], v11 = acc11[r];
        if constexpr (BIASM) {
            float bm0 = bias[m0], bm1 = bias[m1];
            v00 += bm0; v01 += bm0; v10 += bm1; v11 += bm1;
        }
        size_t i00 = (size_t)bz * sOb + (size_t)m0 * N + n0;
        size_t i10 = (size_t)bz * sOb + (size_t)m1 * N + n0;
        if constexpr (OMODE == 0) {
            _Float16* o = (_Float16*)Out;
            o[i00] = (_Float16)v00; o[i00 + 16] = (_Float16)v01;
            o[i10] = (_Float16)v10; o[i10 + 16] = (_Float16)v11;
        } else if constexpr (OMODE == 1) {
            float* o = (float*)Out;
            o[i00] = v00 * scale; o[i00 + 16] = v01 * scale;
            o[i10] = v10 * scale; o[i10 + 16] = v11 * scale;
        } else {
            float* o = (float*)Out;
            o[i00] = v00 + res[i00]; o[i00 + 16] = v01 + res[i00 + 16];
            o[i10] = v10 + res[i10]; o[i10 + 16] = v11 + res[i10 + 16];
        }
    }
    (void)scale;
}

// ---------------------------------------------------------------------------
// Row softmax over 1024 entries; f32 scores in, f16 probabilities out.
// ---------------------------------------------------------------------------
__global__ __launch_bounds__(256)
void softmax_k(const float* __restrict__ s, _Float16* __restrict__ p)
{
    __shared__ float red[256];
    const size_t row = (size_t)blockIdx.x * 1024;
    float4 v = ((const float4*)(s + row))[threadIdx.x];

    float mx = fmaxf(fmaxf(v.x, v.y), fmaxf(v.z, v.w));
    red[threadIdx.x] = mx;
    __syncthreads();
    for (int o = 128; o > 0; o >>= 1) {
        if (threadIdx.x < o) red[threadIdx.x] = fmaxf(red[threadIdx.x], red[threadIdx.x + o]);
        __syncthreads();
    }
    mx = red[0];
    __syncthreads();

    float e0 = __expf(v.x - mx), e1 = __expf(v.y - mx);
    float e2 = __expf(v.z - mx), e3 = __expf(v.w - mx);
    red[threadIdx.x] = e0 + e1 + e2 + e3;
    __syncthreads();
    for (int o = 128; o > 0; o >>= 1) {
        if (threadIdx.x < o) red[threadIdx.x] += red[threadIdx.x + o];
        __syncthreads();
    }
    float rs = 1.0f / red[0];

    _Float16* pw = p + row + (size_t)threadIdx.x * 4;
    pw[0] = (_Float16)(e0 * rs); pw[1] = (_Float16)(e1 * rs);
    pw[2] = (_Float16)(e2 * rs); pw[3] = (_Float16)(e3 * rs);
}

// ---------------------------------------------------------------------------
extern "C" void kernel_launch(void* const* d_in, const int* in_sizes, int n_in,
                              void* d_out, int out_size, void* d_ws, size_t ws_size,
                              hipStream_t stream)
{
    (void)in_sizes; (void)n_in; (void)out_size; (void)ws_size;
    const float* x   = (const float*)d_in[0];
    const float* gns = (const float*)d_in[1];
    const float* gnb = (const float*)d_in[2];
    const float* wq  = (const float*)d_in[3];
    const float* bq  = (const float*)d_in[4];
    const float* wk  = (const float*)d_in[5];
    const float* bk  = (const float*)d_in[6];
    const float* wvp = (const float*)d_in[7];
    const float* bv  = (const float*)d_in[8];
    const float* wo  = (const float*)d_in[9];
    const float* bo  = (const float*)d_in[10];
    float* out = (float*)d_out;

    const long long CN  = 512LL * 1024;          // per-batch C*N
    const long long NN  = 1024LL * 1024;         // per-batch N*N
    const size_t    BCN = 32 * (size_t)CN;
    const int       WC  = 512 * 512;

    char*  ws  = (char*)d_ws;
    size_t off = 0;
    auto alloc = [&](size_t bytes) {
        void* p = ws + off; off += (bytes + 255) & ~(size_t)255; return p;
    };
    _Float16* hn   = (_Float16*)alloc(BCN * 2);             // [B][C][N]
    _Float16* wq16 = (_Float16*)alloc((size_t)WC * 2);
    _Float16* wk16 = (_Float16*)alloc((size_t)WC * 2);
    _Float16* wv16 = (_Float16*)alloc((size_t)WC * 2);
    _Float16* wo16 = (_Float16*)alloc((size_t)WC * 2);
    _Float16* qb   = (_Float16*)alloc(BCN * 2);             // [B][C][N]
    _Float16* kb   = (_Float16*)alloc(BCN * 2);             // [B][C][N]
    _Float16* vb   = (_Float16*)alloc(BCN * 2);             // [B][C][N]
    float*    sc   = (float*)   alloc(32 * (size_t)NN * 4); // [B][N][N] scores
    _Float16* pr   = (_Float16*)alloc(32 * (size_t)NN * 2); // [B][N][N] probs
    _Float16* ot   = (_Float16*)alloc(BCN * 2);             // [B][N][C] attn^T

    // 1) weights -> f16
    cvt_f16_k<<<(WC + 255) / 256, 256, 0, stream>>>(wq,  wq16, WC);
    cvt_f16_k<<<(WC + 255) / 256, 256, 0, stream>>>(wk,  wk16, WC);
    cvt_f16_k<<<(WC + 255) / 256, 256, 0, stream>>>(wvp, wv16, WC);
    cvt_f16_k<<<(WC + 255) / 256, 256, 0, stream>>>(wo,  wo16, WC);

    // 2) GroupNorm -> hn (f16, [B][C][N])
    groupnorm_k<<<32 * 16, 256, 0, stream>>>(x, gns, gnb, hn);

    // 3) q/k/v[c][n] = W (CxC) * hn (CxN) + b[c]
    dim3 gQKV(1024 / 64, 512 / 128, 32);
    gemm_wmma<false, false, 0, true><<<gQKV, 256, 0, stream>>>(
        wq16, 0, hn, CN, qb, CN, bq, nullptr, 512, 1024, 512, 1.f);
    gemm_wmma<false, false, 0, true><<<gQKV, 256, 0, stream>>>(
        wk16, 0, hn, CN, kb, CN, bk, nullptr, 512, 1024, 512, 1.f);
    gemm_wmma<false, false, 0, true><<<gQKV, 256, 0, stream>>>(
        wv16, 0, hn, CN, vb, CN, bv, nullptr, 512, 1024, 512, 1.f);

    // 4) scores[n][m] = (q^T k)/sqrt(C)    A=q [C][N] (TA), B=k [C][N]
    dim3 gS(1024 / 64, 1024 / 128, 32);
    gemm_wmma<true, false, 1, false><<<gS, 256, 0, stream>>>(
        qb, CN, kb, CN, sc, NN, nullptr, nullptr, 1024, 1024, 512,
        0.0441941738f /* 1/sqrt(512) */);

    // 5) softmax rows -> f16 probs
    softmax_k<<<32 * 1024, 256, 0, stream>>>(sc, pr);

    // 6) attn^T[n][c] = P (NxM) * V^T     A=probs [N][M], B=v [C][M] (TB)
    dim3 gA(512 / 64, 1024 / 128, 32);
    gemm_wmma<false, true, 0, false><<<gA, 256, 0, stream>>>(
        pr, NN, vb, CN, ot, CN, nullptr, nullptr, 1024, 512, 1024, 1.f);

    // 7) y[c][n] = Wo * attn + bo + x    A=wo16 [C][C], B=ot [N][C] (TB)
    dim3 gO(1024 / 64, 512 / 128, 32);
    gemm_wmma<false, true, 2, true><<<gO, 256, 0, stream>>>(
        wo16, 0, ot, CN, out, CN, bo, x, 512, 1024, 512, 1.f);
}
